// GRU_36438502539594
// MI455X (gfx1250) — matrix operand order repaired and checked by reading, not compiled, not run on hardware
//
#include <hip/hip_runtime.h>
#include <hip/hip_bf16.h>

// Problem dims (fixed by the reference)
#define B_   512
#define T_   128
#define DIN  256
#define H_   512

#define PANEL_K   256            // K-elements staged per panel
#define LDS_ROW   264            // 256 elems + 8-elem (16B) TDM pad -> bank-conflict-free
#define NGATE     3

typedef __attribute__((ext_vector_type(16))) __bf16         v16bf;
typedef __attribute__((ext_vector_type(8)))  float          v8f;
typedef __attribute__((ext_vector_type(8)))  unsigned short v8us;
typedef __attribute__((ext_vector_type(4)))  unsigned int   v4u;
typedef __attribute__((ext_vector_type(8)))  int            v8i;
typedef __attribute__((ext_vector_type(4)))  int            v4i;

union FragAB { v8us u[2]; v16bf f; };

__device__ __forceinline__ unsigned short f2bf(float x) {
  unsigned int u = __float_as_uint(x);
  u += 0x7FFFu + ((u >> 16) & 1u);      // round-to-nearest-even
  return (unsigned short)(u >> 16);
}

// ---------------------------------------------------------------------------
// One-time converters
// ---------------------------------------------------------------------------
__global__ void cvt_f32_to_bf16(const float* __restrict__ src,
                                unsigned short* __restrict__ dst, int n) {
  int i = blockIdx.x * blockDim.x + threadIdx.x;
  int stride = gridDim.x * blockDim.x;
  for (; i < n; i += stride) dst[i] = f2bf(src[i]);
}

// h [B, 2, H] -> fp32 master state (d_out h_n region) + bf16 copies (buffer 0)
__global__ void init_state(const float* __restrict__ h,
                           float* __restrict__ hn,
                           unsigned short* __restrict__ h0bf,
                           unsigned short* __restrict__ h1bf) {
  int i = blockIdx.x * blockDim.x + threadIdx.x;
  int stride = gridDim.x * blockDim.x;
  for (; i < B_ * H_; i += stride) {
    int b = i / H_, j = i - b * H_;
    float v0 = h[(size_t)b * 2 * H_ + j];
    float v1 = h[(size_t)b * 2 * H_ + H_ + j];
    hn[(size_t)b * 2 * H_ + j]      = v0;
    hn[(size_t)b * 2 * H_ + H_ + j] = v1;
    h0bf[i] = f2bf(v0);
    h1bf[i] = f2bf(v1);
  }
}

// ---------------------------------------------------------------------------
// Fragment loaders (layouts per CDNA5 ISA 7.12.2, 16-bit A 16x32 / B 32x16)
// ---------------------------------------------------------------------------
// A tile 16x32 (M x K): lane l (l15 = l&15) holds row M = l15.
//   lanes 0-15 : elems 0..7 = K[k0+0..7],  elems 8..15 = K[k0+16..23]
//   lanes 16-31: elems 0..7 = K[k0+8..15], elems 8..15 = K[k0+24..31]
__device__ __forceinline__ FragAB load_a(const unsigned short* __restrict__ base,
                                         int ld, int k0, int l15, int lhi) {
  FragAB fr;
  const unsigned short* p = base + (size_t)l15 * ld + k0 + (lhi ? 8 : 0);
  fr.u[0] = *reinterpret_cast<const v8us*>(p);
  fr.u[1] = *reinterpret_cast<const v8us*>(p + 16);
  return fr;
}

// B tile 32x16 (K x N) read from an LDS-staged weight panel (row-major
// 16 rows x PANEL_K elems, LDS_ROW stride). lane l holds column N = l&15;
// lanes 0-15: K = kk+0..7, kk+16..23 ; lanes 16-31: K = kk+8..15, kk+24..31.
__device__ __forceinline__ FragAB load_b_lds(const unsigned short* base,
                                             int kk, int l15, int lhi) {
  FragAB fr;
  const unsigned short* p = base + l15 * LDS_ROW + kk + (lhi ? 8 : 0);
  fr.u[0] = *reinterpret_cast<const v8us*>(p);
  fr.u[1] = *reinterpret_cast<const v8us*>(p + 16);
  return fr;
}

#define WMMA_BF16(A, Bm, C) \
  __builtin_amdgcn_wmma_f32_16x16x32_bf16(false, (A), false, (Bm), (short)0, (C), false, false)

// ---------------------------------------------------------------------------
// TDM: DMA one 16-row x PANEL_K 2-byte tile of a row-major weight matrix
// into LDS, inserting 16B of pad per 512B row (LDS row stride = 528B).
// D# layout per CDNA5 ISA ch.8 (group0 128b, group1 256b, groups 2/3 = 0).
// ---------------------------------------------------------------------------
__device__ __forceinline__ void tdm_load_tile(const unsigned short* gsrc,
                                              int ld_elems,
                                              unsigned lds_byte_addr) {
  unsigned long long ga = (unsigned long long)(uintptr_t)gsrc;
  v4u g0;
  g0[0] = 1u;                                  // count=1, user descriptor
  g0[1] = lds_byte_addr;                       // lds_addr [63:32]
  g0[2] = (unsigned)ga;                        // global_addr lo
  g0[3] = ((unsigned)(ga >> 32) & 0x01FFFFFFu) // global_addr [56:32]
          | 0x80000000u;                       // type=2 ("image") in [127:126]
  v8i g1;
  g1[0] = (1 << 16)                            // data_size = 1 -> 2 bytes
        | (1 << 20)                            // pad_enable
        | (6 << 22)                            // pad_interval: 128 DWORDs (512B)
        | (3 << 25);                           // pad_amount: 4 DWORDs (16B)
  g1[1] = (ld_elems & 0xFFFF) << 16;           // tensor_dim0 lo16  (bits 63:48)
  g1[2] = ((unsigned)ld_elems >> 16)           // tensor_dim0 hi16
        | ((3 * H_) & 0xFFFF) << 16;           // tensor_dim1 lo16
  g1[3] = ((3 * H_) >> 16)                     // tensor_dim1 hi16
        | (PANEL_K << 16);                     // tile_dim0 = 256
  g1[4] = 16;                                  // tile_dim1 = 16 rows, tile_dim2 = 0
  g1[5] = ld_elems;                            // tensor_dim0_stride lo32
  g1[6] = 0;
  g1[7] = 0;
  v4i gz = {0, 0, 0, 0};
#if __clang_major__ >= 23
  v8i gz8 = {0, 0, 0, 0, 0, 0, 0, 0};
  __builtin_amdgcn_tensor_load_to_lds(g0, g1, gz, gz, gz8, 0);
#else
  __builtin_amdgcn_tensor_load_to_lds(g0, g1, gz, gz, 0);
#endif
}

// ---------------------------------------------------------------------------
// One GRU timestep for one layer.
//   r,z accumulate over K_x (input) + K_h (hidden); n uses split accumulators
//   (i_n over x, h_n over hidden):  n = tanh(i_n + r * h_n_term).
// Block = 256 threads = 8 wave32, all sharing one 16-col j-tile;
// WG tile = 128(b) x 16(j).  Grid = (B/128, H/16) = (4, 32).
// Weight panels (3 gates x 16 rows x 256 K) staged in LDS by the TDM,
// double-buffered; A fragments are per-wave direct global loads.
// ---------------------------------------------------------------------------
__global__ __launch_bounds__(256)
void gru_step(const unsigned short* __restrict__ xbf, int ldx, int Kx,
              const unsigned short* __restrict__ hbf_in,   // [B,H] bf16 prev state
              const unsigned short* __restrict__ wi,       // [3H,Kx] bf16
              const unsigned short* __restrict__ wh,       // [3H,H]  bf16
              const float* __restrict__ b_ih,
              const float* __restrict__ b_hh,
              const unsigned char* __restrict__ init_t,    // is_init[b,t], stride T_
              float* __restrict__ hstate,                  // fp32 state, row stride 2H
              unsigned short* __restrict__ hbf_out,        // [B,H] bf16 new state
              float* __restrict__ outp,                    // output slice, row stride T*H
              int write_out) {
  __shared__ __align__(16) unsigned short lds[2][NGATE][16 * LDS_ROW]; // ~50.7KB

  const int lane = threadIdx.x & 31;
  const int wave = threadIdx.x >> 5;
  const int l15  = lane & 15;
  const int lhi  = lane >> 4;
  const int b0 = blockIdx.x * 128 + wave * 16;
  const int j0 = blockIdx.y * 16;

  const int Px = Kx / PANEL_K;          // panels from w_ih (A = x)
  const int P  = Px + H_ / PANEL_K;     // + panels from w_hh (A = h_prev)

  v8f acc_r = {}, acc_z = {}, acc_ni = {}, acc_nh = {};
  const v8us vz = {0, 0, 0, 0, 0, 0, 0, 0};
  const bool rstA = init_t[(size_t)(b0 + l15) * T_] != 0;

  // --- TDM panel issue: 3 tiles (one per gate) into lds[buf] ---
  auto issue_panel = [&](int p, int buf) {
    const unsigned short* wsrc;
    int k0, ld;
    if (p < Px) { wsrc = wi; k0 = p * PANEL_K;        ld = Kx; }
    else        { wsrc = wh; k0 = (p - Px) * PANEL_K; ld = H_; }
#pragma unroll
    for (int g = 0; g < NGATE; ++g) {
      const unsigned short* gp = wsrc + (size_t)(g * H_ + j0) * ld + k0;
      tdm_load_tile(gp, ld, (unsigned)(uintptr_t)&lds[buf][g][0]);
    }
  };

  if (wave == 0) issue_panel(0, 0);

  for (int p = 0; p < P; ++p) {
    const int cur = p & 1;
    if (wave == 0) {
      if (p + 1 < P) {
        issue_panel(p + 1, cur ^ 1);
        __builtin_amdgcn_s_wait_tensorcnt(NGATE);  // panel p done; p+1 in flight
      } else {
        __builtin_amdgcn_s_wait_tensorcnt(0);      // last panel: drain
      }
    }
    __syncthreads();                               // panel p visible to all waves

    // A source for this panel
    const unsigned short* abase;
    int lda, koff;
    bool rst, is_x;
    if (p < Px) { abase = xbf    + (size_t)b0 * ldx; lda = ldx; koff = p * PANEL_K;        rst = false; is_x = true; }
    else        { abase = hbf_in + (size_t)b0 * H_;  lda = H_;  koff = (p - Px) * PANEL_K; rst = rstA;  is_x = false; }

#pragma unroll
    for (int kk = 0; kk < PANEL_K; kk += 32) {
      FragAB a = load_a(abase, lda, koff + kk, l15, lhi);
      if (rst) { a.u[0] = vz; a.u[1] = vz; }
      FragAB br = load_b_lds(&lds[cur][0][0], kk, l15, lhi);
      FragAB bz = load_b_lds(&lds[cur][1][0], kk, l15, lhi);
      FragAB bn = load_b_lds(&lds[cur][2][0], kk, l15, lhi);
      acc_r = WMMA_BF16(a.f, br.f, acc_r);
      acc_z = WMMA_BF16(a.f, bz.f, acc_z);
      if (is_x) acc_ni = WMMA_BF16(a.f, bn.f, acc_ni);
      else      acc_nh = WMMA_BF16(a.f, bn.f, acc_nh);
    }
    __syncthreads();                               // all waves done with lds[cur]
  }

  // ---- epilogue: gates + state blend. C layout: elem e -> M = e + 8*lhi,
  //      N = lane&15 (column j fixed per lane -> biases loaded once).
  const int j = j0 + l15;
  const float bir = b_ih[j],          bhr = b_hh[j];
  const float biz = b_ih[H_ + j],     bhz = b_hh[H_ + j];
  const float bin = b_ih[2 * H_ + j], bhn = b_hh[2 * H_ + j];
#pragma unroll
  for (int e = 0; e < 8; ++e) {
    const int row = b0 + e + lhi * 8;
    const bool rst = init_t[(size_t)row * T_] != 0;
    const float hprev = rst ? 0.0f : hstate[(size_t)row * (2 * H_) + j];
    const float r = 1.0f / (1.0f + __expf(-(acc_r[e] + bir + bhr)));
    const float z = 1.0f / (1.0f + __expf(-(acc_z[e] + biz + bhz)));
    const float n = tanhf(acc_ni[e] + bin + r * (acc_nh[e] + bhn));
    const float hnew = (1.0f - z) * n + z * hprev;
    hstate[(size_t)row * (2 * H_) + j] = hnew;
    hbf_out[(size_t)row * H_ + j] = f2bf(hnew);
    if (write_out) outp[(size_t)row * ((size_t)T_ * H_) + j] = hnew;
  }
}

// ---------------------------------------------------------------------------
extern "C" void kernel_launch(void* const* d_in, const int* in_sizes, int n_in,
                              void* d_out, int out_size, void* d_ws, size_t ws_size,
                              hipStream_t stream) {
  (void)in_sizes; (void)n_in; (void)out_size; (void)ws_size;
  const float*         x       = (const float*)d_in[0];
  const unsigned char* is_init = (const unsigned char*)d_in[1];  // bool8
  const float*         h       = (const float*)d_in[2];
  const float* w_ih0 = (const float*)d_in[3];
  const float* w_hh0 = (const float*)d_in[4];
  const float* b_ih0 = (const float*)d_in[5];
  const float* b_hh0 = (const float*)d_in[6];
  const float* w_ih1 = (const float*)d_in[7];
  const float* w_hh1 = (const float*)d_in[8];
  const float* b_ih1 = (const float*)d_in[9];
  const float* b_hh1 = (const float*)d_in[10];

  float* out = (float*)d_out;                      // [B,T,H]
  float* hn  = out + (size_t)B_ * T_ * H_;         // [B,2,H] fp32 master state

  // Workspace carve-up (bf16 shadows; ~40 MB)
  unsigned short* ws  = (unsigned short*)d_ws;
  size_t off = 0;
  unsigned short* xbf = ws + off; off += (size_t)B_ * T_ * DIN;
  unsigned short* w0i = ws + off; off += (size_t)3 * H_ * DIN;
  unsigned short* w0h = ws + off; off += (size_t)3 * H_ * H_;
  unsigned short* w1i = ws + off; off += (size_t)3 * H_ * H_;
  unsigned short* w1h = ws + off; off += (size_t)3 * H_ * H_;
  unsigned short* h0bf[2] = { ws + off, ws + off + (size_t)B_ * H_ };
  off += 2 * (size_t)B_ * H_;
  unsigned short* h1bf[2] = { ws + off, ws + off + (size_t)B_ * H_ };
  off += 2 * (size_t)B_ * H_;

  cvt_f32_to_bf16<<<2048, 256, 0, stream>>>(x,     xbf, B_ * T_ * DIN);
  cvt_f32_to_bf16<<<512,  256, 0, stream>>>(w_ih0, w0i, 3 * H_ * DIN);
  cvt_f32_to_bf16<<<512,  256, 0, stream>>>(w_hh0, w0h, 3 * H_ * H_);
  cvt_f32_to_bf16<<<512,  256, 0, stream>>>(w_ih1, w1i, 3 * H_ * H_);
  cvt_f32_to_bf16<<<512,  256, 0, stream>>>(w_hh1, w1h, 3 * H_ * H_);
  init_state<<<512, 256, 0, stream>>>(h, hn, h0bf[0], h1bf[0]);

  dim3 grid(B_ / 128, H_ / 16);
  for (int t = 0; t < T_; ++t) {
    const int bi = t & 1, bo = bi ^ 1;
    // layer 0: x slice (row stride T*DIN), K_x = DIN
    gru_step<<<grid, 256, 0, stream>>>(
        xbf + (size_t)t * DIN, T_ * DIN, DIN,
        h0bf[bi], w0i, w0h, b_ih0, b_hh0, is_init + t,
        hn, h0bf[bo], (float*)nullptr, 0);
    // layer 1: x = fresh layer-0 output (row stride H), K_x = H
    gru_step<<<grid, 256, 0, stream>>>(
        h0bf[bo], H_, H_,
        h1bf[bi], w1i, w1h, b_ih1, b_hh1, is_init + t,
        hn + H_, h1bf[bo], out + (size_t)t * H_, 1);
  }
}
